// CollectionEdgeScorer_28905129902361
// MI455X (gfx1250) — matrix operand order repaired and checked by reading, not compiled, not run on hardware
//
#include <hip/hip_runtime.h>
#include <hip/hip_bf16.h>
#include <math.h>

typedef _Float16 h8   __attribute__((ext_vector_type(8)));
typedef _Float16 v16h __attribute__((ext_vector_type(16)));
typedef float    v8f  __attribute__((ext_vector_type(8)));

#define B_ 2
#define S_ 512
#define T_ 512
#define D_ 256
#define H_ 128

// Exact-gelu via branchless Abramowitz-Stegun 7.1.26 erf (|err| < 1.5e-7).
// Uses v_rcp_f32 / v_exp_f32 (TRANS, co-executes with VALU) + ~10 FMAs,
// vs ~25 VALU ops for the ocml erff call.
__device__ __forceinline__ float gelu_exact(float x) {
    float z  = x * 0.70710678118654752f;     // x / sqrt(2)
    float az = fabsf(z);
    float t  = __builtin_amdgcn_rcpf(__builtin_fmaf(0.3275911f, az, 1.0f));
    float p  = 1.061405429f;
    p = __builtin_fmaf(p, t, -1.453152027f);
    p = __builtin_fmaf(p, t,  1.421413741f);
    p = __builtin_fmaf(p, t, -0.284496736f);
    p = __builtin_fmaf(p, t,  0.254829592f);
    p = p * t;
    float e  = __builtin_amdgcn_exp2f(az * az * -1.4426950408889634f);
    float er = __builtin_fmaf(-p, e, 1.0f);  // erf(|z|)
    er = copysignf(er, z);
    return 0.5f * x * (1.0f + er);
}

// ---------------- K0: transpose W1_st (rows 2H..3H of W1) into f16 Wt[k_out][h] ---------
__global__ void k_wt(const float* __restrict__ W1, _Float16* __restrict__ Wt) {
    int idx = blockIdx.x * 256 + threadIdx.x;      // < H_*H_
    int n = idx >> 7;                              // output column k
    int h = idx & (H_ - 1);                        // contraction index
    Wt[idx] = (_Float16)W1[(2 * H_ + h) * H_ + n];
}

// ---------------- K1: fused LayerNorm + projection (source and target) ------------------
__global__ void k_ln_proj(const float* __restrict__ src, const float* __restrict__ tgt,
                          const float* __restrict__ sg, const float* __restrict__ sbb,
                          const float* __restrict__ tg, const float* __restrict__ tbb,
                          const float* __restrict__ Wsu, const float* __restrict__ bsu,
                          const float* __restrict__ Wtp, const float* __restrict__ btp,
                          float* __restrict__ proj_s, float* __restrict__ proj_t,
                          _Float16* __restrict__ hs, _Float16* __restrict__ ht) {
    int row = blockIdx.x;          // 0 .. B*(S+T)-1
    int tid = threadIdx.x;         // 0 .. 255 (== D_)
    bool isSrc = row < B_ * S_;
    const float *x, *g, *bb, *W, *bias;
    float* pf; _Float16* ph;
    if (isSrc) { int r = row;           x = src + r * D_; g = sg; bb = sbb; W = Wsu; bias = bsu;
                 pf = proj_s + r * H_;  ph = hs + r * H_; }
    else       { int r = row - B_ * S_; x = tgt + r * D_; g = tg; bb = tbb; W = Wtp; bias = btp;
                 pf = proj_t + r * H_;  ph = ht + r * H_; }

    __shared__ float red[256];
    __shared__ float xn[D_];
    float xi = x[tid];
    red[tid] = xi; __syncthreads();
    for (int s = 128; s > 0; s >>= 1) { if (tid < s) red[tid] += red[tid + s]; __syncthreads(); }
    float m = red[0] * (1.0f / D_); __syncthreads();
    float d = xi - m;
    red[tid] = d * d; __syncthreads();
    for (int s = 128; s > 0; s >>= 1) { if (tid < s) red[tid] += red[tid + s]; __syncthreads(); }
    float var = red[0] * (1.0f / D_);
    float inv = rsqrtf(var + 1e-5f);
    xn[tid] = d * inv * g[tid] + bb[tid];
    __syncthreads();
    if (tid < H_) {
        float acc = bias[tid];
        #pragma unroll 8
        for (int dd = 0; dd < D_; ++dd) acc += xn[dd] * W[dd * H_ + tid];
        pf[tid] = acc;
        ph[tid] = (_Float16)acc;
    }
}

// ---------------- K2: u = src_proj @ W1_s + b1 ; v = tgt_proj @ W1_t --------------------
__global__ void k_uv(const float* __restrict__ proj_s, const float* __restrict__ proj_t,
                     const float* __restrict__ W1, const float* __restrict__ b1,
                     float* __restrict__ u, float* __restrict__ v) {
    int row = blockIdx.x;          // 0 .. B*(S+T)-1
    int k = threadIdx.x;           // 0 .. 127
    __shared__ float p[H_];
    bool isSrc = row < B_ * S_;
    const float* pr = isSrc ? (proj_s + row * H_) : (proj_t + (row - B_ * S_) * H_);
    p[k] = pr[k]; __syncthreads();
    const float* Wp = isSrc ? W1 : (W1 + H_ * H_);
    float acc = isSrc ? b1[k] : 0.0f;
    #pragma unroll 8
    for (int h = 0; h < H_; ++h) acc += p[h] * Wp[h * H_ + k];
    if (isSrc) u[row * H_ + k] = acc;
    else       v[(row - B_ * S_) * H_ + k] = acc;
}

// ---------------- K3: fused pairwise WMMA + gelu + W2 reduce + softsign -----------------
__global__ void __launch_bounds__(256)
k_edge(const _Float16* __restrict__ hs, const _Float16* __restrict__ ht,
       const _Float16* __restrict__ Wt, const float* __restrict__ u,
       const float* __restrict__ v, const float* __restrict__ W2,
       const float* __restrict__ b2, const unsigned char* __restrict__ mask,
       float* __restrict__ out) {
    int tid  = threadIdx.x;
    int lane = tid & 31;
    int wave = tid >> 5;
    int tile = blockIdx.x * 8 + wave;      // B_*S_*(T_/16) tiles exactly
    int tIdx = tile & 31;                  // T_/16 == 32
    int s    = (tile >> 5) & (S_ - 1);
    int b    = tile >> 14;                 // / (32*512)
    int t0   = tIdx * 16;

    int mrow = lane & 15;                  // A row (t offset) / B column (n local)
    int hi   = lane >> 4;                  // lane half
    int kbaseA = hi * 8;
    int kbaseB = hi * 8;

    const _Float16* srow = hs + ((b * S_ + s) * H_);
    const _Float16* trow = ht + ((b * T_ + t0 + mrow) * H_);
    const float*    urow = u + (b * S_ + s) * H_;
    const float*    vb   = v + (b * T_ + t0 + hi * 8) * H_;

    // Preload C = u[s,col] + v[t,col] into the WMMA accumulators: the first
    // WMMA folds the bias-planes add in for free and the loads overlap math.
    v8f acc[8];
    float w2v[8];
    #pragma unroll
    for (int nt = 0; nt < 8; ++nt) {
        int col = nt * 16 + mrow;
        float uv = urow[col];
        w2v[nt]  = W2[col];
        #pragma unroll
        for (int r = 0; r < 8; ++r)
            acc[nt][r] = uv + vb[r * H_ + col];
    }

    #pragma unroll
    for (int ks = 0; ks < 4; ++ks) {
        int ka = ks * 32 + kbaseA;
        h8 ta0 = *(const h8*)(trow + ka);
        h8 ta1 = *(const h8*)(trow + ka + 16);
        h8 sa0 = *(const h8*)(srow + ka);
        h8 sa1 = *(const h8*)(srow + ka + 16);
        h8 p0 = ta0 * sa0;                 // packed f16: (tgt ⊙ src)
        h8 p1 = ta1 * sa1;
        v16h a = __builtin_shufflevector(p0, p1, 0,1,2,3,4,5,6,7,8,9,10,11,12,13,14,15);
        int kb = ks * 32 + kbaseB;
        #pragma unroll
        for (int nt = 0; nt < 8; ++nt) {
            const _Float16* wp = Wt + (nt * 16 + mrow) * H_ + kb;
            h8 w0 = *(const h8*)(wp);
            h8 w1 = *(const h8*)(wp + 16);
            v16h bf = __builtin_shufflevector(w0, w1, 0,1,2,3,4,5,6,7,8,9,10,11,12,13,14,15);
            acc[nt] = __builtin_amdgcn_wmma_f32_16x16x32_f16(
                false, a, false, bf, (short)0, acc[nt], false, false);
        }
    }

    // gelu + dot with W2, reduce over the 128 k's
    float sum[8];
    #pragma unroll
    for (int r = 0; r < 8; ++r) sum[r] = 0.0f;
    #pragma unroll
    for (int nt = 0; nt < 8; ++nt) {
        #pragma unroll
        for (int r = 0; r < 8; ++r) {
            float gl = gelu_exact(acc[nt][r]);
            sum[r] = __builtin_fmaf(gl, w2v[nt], sum[r]);
        }
    }
    #pragma unroll
    for (int r = 0; r < 8; ++r)
        for (int mm = 1; mm < 16; mm <<= 1)
            sum[r] += __shfl_xor(sum[r], mm, 32);   // reduce within each 16-lane half

    if (mrow == 0) {                                // lane 0 -> rows 0-7, lane 16 -> rows 8-15
        float bias2 = b2[0];
        float mk = mask[b * S_ + s] ? 1.0f : 0.0f;
        float e[8];
        #pragma unroll
        for (int r = 0; r < 8; ++r) {
            float sc = sum[r] + bias2;
            e[r] = (sc / (1.0f + fabsf(sc))) * mk;
        }
        float* op = out + ((size_t)(b * S_ + s)) * T_ + t0 + hi * 8;
        *(float4*)(op)     = make_float4(e[0], e[1], e[2], e[3]);
        *(float4*)(op + 4) = make_float4(e[4], e[5], e[6], e[7]);
    }
}

extern "C" void kernel_launch(void* const* d_in, const int* in_sizes, int n_in,
                              void* d_out, int out_size, void* d_ws, size_t ws_size,
                              hipStream_t stream) {
    (void)in_sizes; (void)n_in; (void)out_size; (void)ws_size;
    const float* src  = (const float*)d_in[0];
    const float* tgt  = (const float*)d_in[1];
    const unsigned char* mask = (const unsigned char*)d_in[2];
    const float* sn_g = (const float*)d_in[3];
    const float* sn_b = (const float*)d_in[4];
    const float* tn_g = (const float*)d_in[5];
    const float* tn_b = (const float*)d_in[6];
    const float* W_su = (const float*)d_in[7];
    const float* b_su = (const float*)d_in[8];
    const float* W_tp = (const float*)d_in[9];
    const float* b_tp = (const float*)d_in[10];
    const float* W1   = (const float*)d_in[11];
    const float* b1   = (const float*)d_in[12];
    const float* W2   = (const float*)d_in[13];
    const float* b2   = (const float*)d_in[14];

    char* ws = (char*)d_ws;
    _Float16* hs = (_Float16*)(ws);                 // B*S*H f16      : 256 KB
    _Float16* ht = (_Float16*)(ws + 262144);        // B*T*H f16      : 256 KB
    _Float16* Wt = (_Float16*)(ws + 524288);        // H*H f16        :  32 KB
    float* u  = (float*)(ws + 557056);              // B*S*H f32      : 512 KB
    float* v  = (float*)(ws + 1081344);             // B*T*H f32      : 512 KB
    float* ps = (float*)(ws + 1605632);             // B*S*H f32      : 512 KB
    float* pt = (float*)(ws + 2129920);             // B*T*H f32      : 512 KB

    k_wt     <<<(H_ * H_) / 256, 256, 0, stream>>>(W1, Wt);
    k_ln_proj<<<B_ * (S_ + T_), 256, 0, stream>>>(src, tgt, sn_g, sn_b, tn_g, tn_b,
                                                  W_su, b_su, W_tp, b_tp, ps, pt, hs, ht);
    k_uv     <<<B_ * (S_ + T_), 128, 0, stream>>>(ps, pt, W1, b1, u, v);
    k_edge   <<<(B_ * S_ * (T_ / 16)) / 8, 256, 0, stream>>>(hs, ht, Wt, u, v, W2, b2, mask,
                                                             (float*)d_out);
}
